// ExtendedS5SSM_45183055954615
// MI455X (gfx1250) — compile-verified, matrix operand order checked by reading, not run on hardware
//
#include <hip/hip_runtime.h>
#include <hip/hip_bf16.h>

// ---------------- problem constants ----------------
#define LSEQ 4096
#define HDIM 1024
#define PDIM 512
#define RDIM 16
#define P2   1024   // 2*P
#define P4   2048   // 4*P
#define NCHUNK 64
#define CLEN   64   // NCHUNK*CLEN == LSEQ

typedef __attribute__((ext_vector_type(16))) __bf16 v16bf;
typedef __attribute__((ext_vector_type(8)))  __bf16 v8bf;
typedef __attribute__((ext_vector_type(8)))  float  v8f;

__device__ __forceinline__ __bf16 f2bf(float x) {
    unsigned u = __builtin_bit_cast(unsigned, x);
    u += 0x7FFFu + ((u >> 16) & 1u);          // round-to-nearest-even
    unsigned short s = (unsigned short)(u >> 16);
    return __builtin_bit_cast(__bf16, s);
}

// ---------------- setup kernels ----------------

// per-state discretization: Lambda_bar = exp(Lam*step), coef = (Lambda_bar-1)/Lam
__global__ void k_prep_state(const float* __restrict__ Lre, const float* __restrict__ Lim,
                             const float* __restrict__ log_step,
                             float2* __restrict__ lbar,   // (P) re,im
                             float2* __restrict__ coef) { // (P)
    int p = blockIdx.x * blockDim.x + threadIdx.x;
    if (p >= PDIM) return;
    float lr = fminf(Lre[p], -1e-4f);
    float li = Lim[p];
    float st = __expf(log_step[p]);
    float mag = __expf(lr * st);
    float ang = li * st;
    float br = mag * __cosf(ang);
    float bi = mag * __sinf(ang);
    lbar[p] = make_float2(br, bi);
    float nr = br - 1.0f, ni = bi;
    float den = lr*lr + li*li;
    coef[p] = make_float2((nr*lr + ni*li) / den, (ni*lr - nr*li) / den);
}

// EF[r,p] = E[p,r]*F[r,p] (complex elementwise), stored (R,P) float2
__global__ void k_prep_EF(const float* __restrict__ Ere, const float* __restrict__ Eim,
                          const float* __restrict__ Fre, const float* __restrict__ Fim,
                          float2* __restrict__ EF) {
    int idx = blockIdx.x * blockDim.x + threadIdx.x;
    if (idx >= RDIM * PDIM) return;
    int r = idx / PDIM, p = idx % PDIM;
    float er = Ere[p*RDIM + r], ei = Eim[p*RDIM + r];
    float fr = Fre[idx],        fi = Fim[idx];
    EF[idx] = make_float2(er*fr - ei*fi, er*fi + ei*fr);
}

// B_bar bf16, stored (2P x H): row 2p -> Re(coef_p * B_tilde[p,:]), row 2p+1 -> Im
// (interleaved so GEMM1 output columns are (re,im) pairs -> float2 loads in scan)
__global__ void k_build_Bbf(const float* __restrict__ Bre, const float* __restrict__ Bim,
                            const float2* __restrict__ coef, __bf16* __restrict__ Bbf) {
    int n = blockIdx.x;
    int p = n >> 1;
    int isIm = n & 1;
    float2 c = coef[p];
    for (int h = threadIdx.x; h < HDIM; h += blockDim.x) {
        float br = Bre[p*HDIM + h], bi = Bim[p*HDIM + h];
        float re = c.x*br - c.y*bi;
        float im = c.x*bi + c.y*br;
        Bbf[n*HDIM + h] = f2bf(isIm ? im : re);
    }
}

// Ct bf16, stored (H x 4P): col k<2P -> 2*C_re[h,k], k>=2P -> -2*C_im[h,k-2P]
__global__ void k_build_Ct(const float* __restrict__ Cre, const float* __restrict__ Cim,
                           __bf16* __restrict__ Ct) {
    int h = blockIdx.x;
    for (int k = threadIdx.x; k < P4; k += blockDim.x) {
        float v = (k < P2) ? (2.0f * Cre[h*P2 + k]) : (-2.0f * Cim[h*P2 + (k - P2)]);
        Ct[h*P4 + k] = f2bf(v);
    }
}

__global__ void k_cvt_u(const float* __restrict__ u, __bf16* __restrict__ ubf, int n) {
    int i = blockIdx.x * blockDim.x + threadIdx.x;
    if (i < n) ubf[i] = f2bf(u[i]);
}

// Lambda_elements[l,p] = Lambda_bar[p] + sum_r Delta[l,r]*EF[r,p], stored (L,P) float2
__global__ void k_lam_elem(const float2* __restrict__ lbar,
                           const float* __restrict__ Delta,
                           const float2* __restrict__ EF,
                           float2* __restrict__ lam) {
    int l = blockIdx.x;
    int p = blockIdx.y * blockDim.x + threadIdx.x;
    float2 a = lbar[p];
    #pragma unroll
    for (int r = 0; r < RDIM; ++r) {
        float d = Delta[l*RDIM + r];
        float2 ef = EF[r*PDIM + p];
        a.x += d * ef.x;
        a.y += d * ef.y;
    }
    lam[l*PDIM + p] = a;
}

// ---------------- bf16 WMMA GEMM ----------------
// C(M x 1024) = A(M x K, row-major bf16) x Bt(1024 x K, row-major bf16)^T
// block = 256 threads = 8 waves (2 along M x 4 along N)
// wave tile 64x32 (4x2 wmma tiles) -> block tile 128x128
// EPI: out += Dvec[col]*U[row*1024+col] (compile-time specialized, no branches)
template <bool EPI>
__global__ __launch_bounds__(256)
void k_gemm_bf16(const __bf16* __restrict__ A, const __bf16* __restrict__ Bt,
                 float* __restrict__ Cout, int K,
                 const float* __restrict__ Dvec, const float* __restrict__ U) {
    const int N = 1024;
    int tid  = threadIdx.x;
    int lane = tid & 31;
    int w    = tid >> 5;
    int wm   = w >> 2;       // 2 waves along M
    int wn   = w & 3;        // 4 waves along N
    int r    = lane & 15;
    int ksel = lane >> 4;    // 0: K {0..7,16..23}, 1: K {8..15,24..31}

    int rowBase = blockIdx.x * 128 + wm * 64;
    int colBase = blockIdx.y * 128 + wn * 32;

    v8f acc[4][2] = {};

    for (int kk = 0; kk < K; kk += 32) {
        v16bf a[4], b[2];
        #pragma unroll
        for (int mi = 0; mi < 4; ++mi) {
            const __bf16* ap = A + (rowBase + mi*16 + r) * K + kk + ksel * 8;
            v8bf lo = *(const v8bf*)ap;
            v8bf hi = *(const v8bf*)(ap + 16);
            a[mi] = __builtin_shufflevector(lo, hi, 0,1,2,3,4,5,6,7,8,9,10,11,12,13,14,15);
        }
        #pragma unroll
        for (int ni = 0; ni < 2; ++ni) {
            const __bf16* bp = Bt + (colBase + ni*16 + r) * K + kk + ksel * 8;
            v8bf lo = *(const v8bf*)bp;
            v8bf hi = *(const v8bf*)(bp + 16);
            b[ni] = __builtin_shufflevector(lo, hi, 0,1,2,3,4,5,6,7,8,9,10,11,12,13,14,15);
        }
        #pragma unroll
        for (int mi = 0; mi < 4; ++mi)
            #pragma unroll
            for (int ni = 0; ni < 2; ++ni)
                acc[mi][ni] = __builtin_amdgcn_wmma_f32_16x16x32_bf16(
                    false, a[mi], false, b[ni], (short)0, acc[mi][ni], false, false);
    }

    // C/D layout: VGPR e -> M = e + 8*ksel, N = lane&15
    #pragma unroll
    for (int mi = 0; mi < 4; ++mi) {
        #pragma unroll
        for (int ni = 0; ni < 2; ++ni) {
            int col = colBase + ni*16 + r;
            #pragma unroll
            for (int e = 0; e < 8; ++e) {
                int row = rowBase + mi*16 + ksel*8 + e;
                float v = acc[mi][ni][e];
                if constexpr (EPI) v += Dvec[col] * U[row * N + col];
                Cout[row * N + col] = v;
            }
        }
    }
}

// ---------------- chunked complex scans ----------------
// agg: (NCHUNK*P) float4 {a_re, a_im, x_re, x_im}; carry: (NCHUNK*P) float2

__global__ void k_scan_phase1(const float2* __restrict__ lam, const float2* __restrict__ bu,
                              float4* __restrict__ agg, int rev) {
    int p = blockIdx.y * blockDim.x + threadIdx.x;
    int c = blockIdx.x;
    int lo = c * CLEN;
    float are = 1.f, aim = 0.f, xre = 0.f, xim = 0.f;
    for (int i = 0; i < CLEN; ++i) {
        int t = rev ? (lo + CLEN - 1 - i) : (lo + i);
        float2 Av = lam[t*PDIM + p];
        float2 bv = bu[t*PDIM + p];
        float nxr = Av.x*xre - Av.y*xim + bv.x;
        float nxi = Av.x*xim + Av.y*xre + bv.y;
        xre = nxr; xim = nxi;
        float nar = Av.x*are - Av.y*aim;
        float nai = Av.x*aim + Av.y*are;
        are = nar; aim = nai;
    }
    agg[c*PDIM + p] = make_float4(are, aim, xre, xim);
}

__global__ void k_scan_phase2(const float4* __restrict__ agg, float2* __restrict__ carry, int rev) {
    int p = blockIdx.x * blockDim.x + threadIdx.x;
    float cr = 0.f, ci = 0.f;
    for (int j = 0; j < NCHUNK; ++j) {
        int c = rev ? (NCHUNK - 1 - j) : j;
        int idx = c*PDIM + p;
        carry[idx] = make_float2(cr, ci);
        float4 g = agg[idx];
        float nr = g.x*cr - g.y*ci + g.z;
        float ni = g.x*ci + g.y*cr + g.w;
        cr = nr; ci = ni;
    }
}

__global__ void k_scan_phase3(const float2* __restrict__ lam, const float2* __restrict__ bu,
                              const float2* __restrict__ carry,
                              __bf16* __restrict__ xs, int rev, int off_re, int off_im) {
    int p = blockIdx.y * blockDim.x + threadIdx.x;
    int c = blockIdx.x;
    int lo = c * CLEN;
    float2 x = carry[c*PDIM + p];
    float xre = x.x, xim = x.y;
    for (int i = 0; i < CLEN; ++i) {
        int t = rev ? (lo + CLEN - 1 - i) : (lo + i);
        float2 Av = lam[t*PDIM + p];
        float2 bv = bu[t*PDIM + p];
        float nxr = Av.x*xre - Av.y*xim + bv.x;
        float nxi = Av.x*xim + Av.y*xre + bv.y;
        xre = nxr; xim = nxi;
        xs[t*P4 + off_re + p] = f2bf(xre);
        xs[t*P4 + off_im + p] = f2bf(xim);
    }
}

// ---------------- launch ----------------
extern "C" void kernel_launch(void* const* d_in, const int* in_sizes, int n_in,
                              void* d_out, int out_size, void* d_ws, size_t ws_size,
                              hipStream_t stream) {
    const float* u     = (const float*)d_in[0];
    const float* Lre   = (const float*)d_in[1];
    const float* Lim   = (const float*)d_in[2];
    const float* Bre   = (const float*)d_in[3];
    const float* Bim   = (const float*)d_in[4];
    const float* Cre   = (const float*)d_in[5];
    const float* Cim   = (const float*)d_in[6];
    const float* Ere   = (const float*)d_in[7];
    const float* Eim   = (const float*)d_in[8];
    const float* Fre   = (const float*)d_in[9];
    const float* Fim   = (const float*)d_in[10];
    const float* Dvec  = (const float*)d_in[11];
    const float* lstep = (const float*)d_in[12];
    const float* Delta = (const float*)d_in[13];
    float* out = (float*)d_out;

    char* ws = (char*)d_ws;
    auto alloc = [&](size_t bytes) -> char* {
        char* pptr = ws;
        ws += (bytes + 255) & ~(size_t)255;
        return pptr;
    };

    __bf16* ubf   = (__bf16*)alloc((size_t)LSEQ * HDIM * 2);
    __bf16* Bbf   = (__bf16*)alloc((size_t)P2 * HDIM * 2);
    __bf16* Ct    = (__bf16*)alloc((size_t)HDIM * P4 * 2);
    float2* Bu    = (float2*)alloc((size_t)LSEQ * PDIM * 8);   // (L,P) complex
    float2* lam   = (float2*)alloc((size_t)LSEQ * PDIM * 8);   // (L,P) complex
    __bf16* xs    = (__bf16*)alloc((size_t)LSEQ * P4 * 2);
    float2* lbar  = (float2*)alloc((size_t)PDIM * 8);
    float2* coef  = (float2*)alloc((size_t)PDIM * 8);
    float2* EF    = (float2*)alloc((size_t)RDIM * PDIM * 8);
    float4* agg   = (float4*)alloc((size_t)NCHUNK * PDIM * 16);
    float2* carry = (float2*)alloc((size_t)NCHUNK * PDIM * 8);

    // setup
    k_prep_state<<<2, 256, 0, stream>>>(Lre, Lim, lstep, lbar, coef);
    k_prep_EF<<<(RDIM*PDIM)/256, 256, 0, stream>>>(Ere, Eim, Fre, Fim, EF);
    k_build_Bbf<<<P2, 256, 0, stream>>>(Bre, Bim, coef, Bbf);
    k_build_Ct<<<HDIM, 256, 0, stream>>>(Cre, Cim, Ct);
    k_cvt_u<<<(LSEQ*HDIM)/256, 256, 0, stream>>>(u, ubf, LSEQ*HDIM);
    k_lam_elem<<<dim3(LSEQ, PDIM/256), 256, 0, stream>>>(lbar, Delta, EF, lam);

    // GEMM 1: Bu(L x 2P, (re,im) interleaved) = ubf(L x H) @ Bbf(2P x H)^T   (K = H)
    k_gemm_bf16<false><<<dim3(LSEQ/128, P2/128), 256, 0, stream>>>(
        ubf, Bbf, (float*)Bu, HDIM, nullptr, nullptr);

    // forward scan -> xs cols [0..P) = re, [2P..3P) = im
    k_scan_phase1<<<dim3(NCHUNK, PDIM/256), 256, 0, stream>>>(lam, Bu, agg, 0);
    k_scan_phase2<<<PDIM/256, 256, 0, stream>>>(agg, carry, 0);
    k_scan_phase3<<<dim3(NCHUNK, PDIM/256), 256, 0, stream>>>(lam, Bu, carry, xs, 0, 0, 2*PDIM);

    // backward scan -> xs cols [P..2P) = re, [3P..4P) = im
    k_scan_phase1<<<dim3(NCHUNK, PDIM/256), 256, 0, stream>>>(lam, Bu, agg, 1);
    k_scan_phase2<<<PDIM/256, 256, 0, stream>>>(agg, carry, 1);
    k_scan_phase3<<<dim3(NCHUNK, PDIM/256), 256, 0, stream>>>(lam, Bu, carry, xs, 1, PDIM, 3*PDIM);

    // GEMM 2: out(L x H) = xs(L x 4P) @ Ct(H x 4P)^T + D*u   (K = 4P)
    k_gemm_bf16<true><<<dim3(LSEQ/128, HDIM/128), 256, 0, stream>>>(
        xs, Ct, out, P4, Dvec, u);
}
